// PG_50672024158615
// MI455X (gfx1250) — matrix operand order repaired and checked by reading, not compiled, not run on hardware
//
#include <hip/hip_runtime.h>

// ---------------- types ----------------
typedef __attribute__((ext_vector_type(16))) _Float16 v16h;
typedef __attribute__((ext_vector_type(8)))  _Float16 v8h;
typedef __attribute__((ext_vector_type(8)))  float    v8f;
typedef __attribute__((address_space(3)))    _Float16 lds_h;

// ---------------- problem constants ----------------
static const int Bn   = 64;      // batch
static const int Sc   = 64;      // module channels
static const int Tn   = 12;      // program length
static const int HWn  = 196;     // 14*14
static const int CDn  = 36864;   // 64*64*9
static const int NAn  = 29;
static const int NSLOT= 45;
// LDS: raw chunk [128][196] halves + im2col panel [32][1152] halves
static const int RAWH = 128 * 196;         // 25088 halves
static const int COLH = 32 * 1152;         // 36864 halves
static const int LDSB = (RAWH + COLH) * 2; // 123904 bytes

// =====================================================================
// helpers
// =====================================================================
__device__ __forceinline__ unsigned lds_byte_off(_Float16* p) {
    // generic -> LDS addrspace -> 32-bit byte offset (operand for async-to-LDS)
    return (unsigned)(size_t)(lds_h*)p;
}

// A (16x32 f16): row-major w[row][K]; lanes 0-15 row=lane K{0..7,16..23},
// lanes 16-31 same rows K{8..15,24..31}. (rowOff+kb) must be 8-half aligned.
__device__ __forceinline__ v16h load_a16(const _Float16* __restrict__ w,
                                         long rowOff, int kb, int lane) {
    int sel = (lane >> 4) & 1;
    const v8h* p = (const v8h*)(w + rowOff + kb);
    v8h lo = p[sel];
    v8h hi = p[sel + 2];
    v16h a;
#pragma unroll
    for (int e = 0; e < 8; ++e) { a[e] = lo[e]; a[e + 8] = hi[e]; }
    return a;
}

// B (32x16 f16): lanes 0-15 K{0..7,16..23}, lanes 16-31 K{8..15,24..31}, N = colOff row.
// two aligned 16B loads (ds_load_b128 when src is LDS).
__device__ __forceinline__ v16h load_b_rm(const _Float16* __restrict__ x,
                                          long colOff, int kb, int lane) {
    int sel = (lane >> 4) & 1;
    const v8h* p = (const v8h*)(x + colOff + kb);
    v8h lo = p[sel];
    v8h hi = p[sel + 2];
    v16h b;
#pragma unroll
    for (int e = 0; e < 8; ++e) { b[e] = lo[e]; b[e + 8] = hi[e]; }
    return b;
}

// =====================================================================
// conv core: per-sample implicit GEMM  C[Cout x 196] = W[Cout x Cin*kk] * im2col
// blockIdx.x = 32-pixel group (7 groups cover 196+pad). block = 128 thr (4 waves).
// Each wave: one m-tile x two n-tiles -> 2 WMMAs per A-fragment load.
// =====================================================================
template<bool BIAS, bool O16, bool O32>
__device__ void conv_core(const _Float16* __restrict__ inA,
                          const _Float16* __restrict__ inB,   // channels 64..127 (binary) or null
                          const _Float16* __restrict__ w,     // [Cout][Cin*kk]
                          const float* __restrict__ bias,
                          _Float16* __restrict__ out16,
                          float* __restrict__ out32,
                          int Cin, int Cout, int k3,
                          _Float16* lds) {
    const int tid  = threadIdx.x;
    const int lane = tid & 31;
    const int wave = tid >> 5;
    const int NW   = blockDim.x >> 5;
    const int pix0 = blockIdx.x * 32;
    const int kmul = k3 ? 9 : 1;
    const int K    = Cin * kmul;
    const int CH   = (Cin <= 128) ? Cin : 128;  // channels per LDS chunk
    const int Kc   = CH * kmul;

    _Float16* raw = lds;            // [CH][196]
    _Float16* col = lds + RAWH;     // [32][Kc] im2col panel
    const unsigned rawB = lds_byte_off(raw);

    // ---- stage one channel chunk into LDS (async copy), then build im2col panel ----
    auto stage = [&](int c0) {
        const int nu = (CH * HWn) >> 3;          // 16-byte units
        const long cbase = (long)c0 * HWn;       // half offset in (concat) input space
        for (int i = tid; i < nu; i += blockDim.x) {
            long g = cbase + ((long)i << 3);
            const _Float16* src = (inB && g >= (long)64 * HWn) ? (inB + (g - 64 * HWn))
                                                               : (inA + g);
            unsigned dst = rawB + ((unsigned)i << 4);
            asm volatile("global_load_async_to_lds_b128 %0, %1, off"
                         :: "v"(dst), "v"((unsigned long long)(size_t)src)
                         : "memory");
        }
        asm volatile("s_wait_asynccnt 0x0" ::: "memory");
        __syncthreads();
        // branch-free im2col: clamped address + select, no exec-mask divergence
#pragma unroll 1
        for (int pl = 0; pl < 32; ++pl) {
            int p = pix0 + pl;
            p = (p < HWn) ? p : 0;               // invalid pixels: junk, never stored
            const int py = p / 14, px = p % 14;
            if (k3) {
                for (int k = tid; k < Kc; k += blockDim.x) {
                    int cl = k / 9, rs = k % 9;
                    int y = py + rs / 3 - 1;
                    int x = px + rs % 3 - 1;
                    bool ok = (y >= 0) & (y < 14) & (x >= 0) & (x < 14);
                    int sidx = cl * HWn + (ok ? (y * 14 + x) : 0);
                    _Float16 v = raw[sidx];
                    col[pl * Kc + k] = ok ? v : (_Float16)0.0f;
                }
            } else {
                for (int k = tid; k < Kc; k += blockDim.x)
                    col[pl * Kc + k] = raw[k * HWn + p];
            }
        }
        __syncthreads();
    };

    auto epilogue = [&](int mt, const v8f& acc, int pixn) {
        if (pixn < HWn) {
#pragma unroll
            for (int r = 0; r < 8; ++r) {
                int m = mt * 16 + r + ((lane & 16) ? 8 : 0);
                float v = acc[r];
                if (BIAS) v += bias[m];
                v = v > 0.0f ? v : 0.0f;   // every conv in this net is ReLU'd
                if (O16) out16[m * HWn + pixn] = (_Float16)v;
                if (O32) out32[m * HWn + pixn] = v;
            }
        }
    };

    const long colOff0 = (long)(lane & 15) * Kc;
    const long colOff1 = (long)((lane & 15) + 16) * Kc;
    const int pixA = pix0 + (lane & 15);
    const int pixB = pixA + 16;

    if (Cin <= 128) {
        // single chunk; im2col panel reused across all m-tiles
        stage(0);
        for (int mt = wave; mt < (Cout >> 4); mt += NW) {
            v8f acc0 = {}, acc1 = {};
            long rowOff = (long)(mt * 16 + (lane & 15)) * K;
#pragma unroll 2
            for (int kb = 0; kb < K; kb += 32) {
                __builtin_prefetch(w + rowOff + kb + 128, 0, 1);
                v16h a  = load_a16(w, rowOff, kb, lane);
                v16h b0 = load_b_rm(col, colOff0, kb, lane);
                v16h b1 = load_b_rm(col, colOff1, kb, lane);
                acc0 = __builtin_amdgcn_wmma_f32_16x16x32_f16(
                    false, a, false, b0, (short)0, acc0, false, false);
                acc1 = __builtin_amdgcn_wmma_f32_16x16x32_f16(
                    false, a, false, b1, (short)0, acc1, false, false);
            }
            epilogue(mt, acc0, pixA);
            epilogue(mt, acc1, pixB);
        }
    } else {
        // chunked over channels (stem conv1, Cin=1024); requires Cout == NW*16
        v8f acc0 = {}, acc1 = {};
        const int mt = wave;
        long rowOff = (long)(mt * 16 + (lane & 15)) * K;
        for (int c0 = 0; c0 < Cin; c0 += 128) {
            stage(c0);
            const int kbase = c0 * kmul;
#pragma unroll 2
            for (int kl = 0; kl < Kc; kl += 32) {
                __builtin_prefetch(w + rowOff + kbase + kl + 128, 0, 1);
                v16h a  = load_a16(w, rowOff, kbase + kl, lane);
                v16h b0 = load_b_rm(col, colOff0, kl, lane);
                v16h b1 = load_b_rm(col, colOff1, kl, lane);
                acc0 = __builtin_amdgcn_wmma_f32_16x16x32_f16(
                    false, a, false, b0, (short)0, acc0, false, false);
                acc1 = __builtin_amdgcn_wmma_f32_16x16x32_f16(
                    false, a, false, b1, (short)0, acc1, false, false);
            }
        }
        epilogue(mt, acc0, pixA);
        epilogue(mt, acc1, pixB);
    }
}

// =====================================================================
// kernels
// =====================================================================
__global__ void k_convert(const float* __restrict__ src, _Float16* __restrict__ dst, int n) {
    for (int i = blockIdx.x * blockDim.x + threadIdx.x; i < n; i += gridDim.x * blockDim.x)
        dst[i] = (_Float16)src[i];
}

// stem convs: bias + f16 output
__global__ void k_conv_stem(const _Float16* __restrict__ in, long inStride,
                            const _Float16* __restrict__ w, const float* __restrict__ bias,
                            _Float16* __restrict__ out16, long oStride,
                            int Cin, int Cout) {
    extern __shared__ _Float16 lds[];
    const long b = blockIdx.y;
    conv_core<true, true, false>(in + b * inStride, nullptr, w, bias,
                                 out16 + b * oStride, nullptr, Cin, Cout, 1, lds);
}

// classifier 1x1 conv: bias + f32 output
__global__ void k_conv_cls(const _Float16* __restrict__ in, long inStride,
                           const _Float16* __restrict__ w, const float* __restrict__ bias,
                           float* __restrict__ out32, long oStride,
                           int Cin, int Cout) {
    extern __shared__ _Float16 lds[];
    const long b = blockIdx.y;
    conv_core<true, false, true>(in + b * inStride, nullptr, w, bias,
                                 nullptr, out32 + b * oStride, Cin, Cout, 0, lds);
}

__global__ void k_init_state(const _Float16* __restrict__ f,
                             _Float16* __restrict__ outA, _Float16* __restrict__ savA, int n) {
    for (int i = blockIdx.x * blockDim.x + threadIdx.x; i < n; i += gridDim.x * blockDim.x) {
        _Float16 v = f[i];
        outA[i] = v; savA[i] = v;
    }
}

__global__ void k_update_saved(const int* __restrict__ progs, int t,
                               const _Float16* __restrict__ outCur,
                               const _Float16* __restrict__ savCur,
                               _Float16* __restrict__ savNext) {
    const int per = Sc * HWn;
    const int n = Bn * per;
    for (int i = blockIdx.x * blockDim.x + threadIdx.x; i < n; i += gridDim.x * blockDim.x) {
        int b = i / per;
        int slot = progs[b * Tn + t];
        savNext[i] = (slot == 4) ? outCur[i] : savCur[i];
    }
}

__global__ void k_prog_conv1(const int* __restrict__ progs, int t,
                             const _Float16* __restrict__ featb,
                             const _Float16* __restrict__ outCur,
                             const _Float16* __restrict__ savCur,
                             const _Float16* __restrict__ memf,
                             _Float16* __restrict__ tmp1) {
    extern __shared__ _Float16 lds[];
    const int b = blockIdx.y;
    const int i = progs[b * Tn + t];
    if (i < 4) return;   // skipped step; propagation handled in k_prog_conv3
    const size_t boff = (size_t)b * (Sc * HWn);
    const _Float16* w = memf + (size_t)i * (4 * CDn);
    if (i == 4)
        conv_core<false, true, false>(featb + boff, nullptr, w, nullptr,
                                      tmp1 + boff, nullptr, 64, 64, 1, lds);
    else if (i < 10)
        conv_core<false, true, false>(outCur + boff, savCur + boff, w, nullptr,
                                      tmp1 + boff, nullptr, 128, 64, 1, lds);
    else
        conv_core<false, true, false>(outCur + boff, nullptr, w, nullptr,
                                      tmp1 + boff, nullptr, 64, 64, 1, lds);
}

__global__ void k_prog_conv2(const int* __restrict__ progs, int t,
                             const _Float16* __restrict__ tmp1,
                             const _Float16* __restrict__ memf,
                             _Float16* __restrict__ tmp2) {
    extern __shared__ _Float16 lds[];
    const int b = blockIdx.y;
    const int i = progs[b * Tn + t];
    if (i < 4) return;
    const size_t boff = (size_t)b * (Sc * HWn);
    const bool binary = (i >= 5 && i < 10);
    const _Float16* w = memf + (size_t)i * (4 * CDn) + (binary ? 2 * CDn : CDn);
    conv_core<false, true, false>(tmp1 + boff, nullptr, w, nullptr,
                                  tmp2 + boff, nullptr, 64, 64, 1, lds);
}

__global__ void k_prog_conv3(const int* __restrict__ progs, int t,
                             const _Float16* __restrict__ tmp2,
                             const _Float16* __restrict__ outCur,
                             const _Float16* __restrict__ memf,
                             _Float16* __restrict__ outNext) {
    extern __shared__ _Float16 lds[];
    const int b = blockIdx.y;
    const int i = progs[b * Tn + t];
    const size_t boff = (size_t)b * (Sc * HWn);
    if (i >= 5 && i < 10) {
        const _Float16* w = memf + (size_t)i * (4 * CDn) + 3 * CDn;
        conv_core<false, true, false>(tmp2 + boff, nullptr, w, nullptr,
                                      outNext + boff, nullptr, 64, 64, 1, lds);
    } else {
        const _Float16* src = (i < 4) ? (outCur + boff) : (tmp2 + boff);
        const int n = Sc * HWn;
        for (int idx = blockIdx.x * blockDim.x + threadIdx.x; idx < n;
             idx += gridDim.x * blockDim.x)
            outNext[boff + idx] = src[idx];
    }
}

__global__ void k_pool(const float* __restrict__ cf, _Float16* __restrict__ x) {
    // cf: [B][512][196] (ReLU'd), x: [B][25088] with feature = c*49 + p
    const int n = Bn * 512 * 49;
    for (int i = blockIdx.x * blockDim.x + threadIdx.x; i < n; i += gridDim.x * blockDim.x) {
        int p = i % 49;
        int c = (i / 49) % 512;
        int b = i / (49 * 512);
        int py = p / 7, px = p % 7;
        const float* base = cf + ((size_t)b * 512 + c) * HWn + (py * 2) * 14 + px * 2;
        float m = fmaxf(fmaxf(base[0], base[1]), fmaxf(base[14], base[15]));
        x[(size_t)b * 25088 + c * 49 + p] = (_Float16)m;
    }
}

// FC1: C[1024 x 64] = W1[1024 x 25088] * X^T ; grid (2 n-pairs, 16 mgroups), 128 thr
// Each wave: one m-tile x two n-tiles.
__global__ void k_fc1(const _Float16* __restrict__ w1, const _Float16* __restrict__ x,
                      const float* __restrict__ b1, float* __restrict__ out) {
    const int lane = threadIdx.x & 31;
    const int wave = threadIdx.x >> 5;
    const int mt = blockIdx.y * 4 + wave;
    const int n0 = blockIdx.x * 32;
    const int K = 25088;
    long rowOff  = (long)(mt * 16 + (lane & 15)) * K;
    long colOff0 = (long)(n0 + (lane & 15)) * K;
    long colOff1 = (long)(n0 + (lane & 15) + 16) * K;
    v8f acc0 = {}, acc1 = {};
#pragma unroll 2
    for (int kb = 0; kb < K; kb += 32) {
        __builtin_prefetch(w1 + rowOff + kb + 256, 0, 1);
        v16h a  = load_a16(w1, rowOff, kb, lane);
        v16h b0 = load_b_rm(x, colOff0, kb, lane);
        v16h b1 = load_b_rm(x, colOff1, kb, lane);
        acc0 = __builtin_amdgcn_wmma_f32_16x16x32_f16(
            false, a, false, b0, (short)0, acc0, false, false);
        acc1 = __builtin_amdgcn_wmma_f32_16x16x32_f16(
            false, a, false, b1, (short)0, acc1, false, false);
    }
#pragma unroll
    for (int r = 0; r < 8; ++r) {
        int m = mt * 16 + r + ((lane & 16) ? 8 : 0);
        float bv = b1[m];
        int na = n0 + (lane & 15);
        float v0 = acc0[r] + bv;
        float v1 = acc1[r] + bv;
        v0 = v0 > 0.0f ? v0 : 0.0f;
        v1 = v1 > 0.0f ? v1 : 0.0f;
        out[na * 1024 + m] = v0;
        out[(na + 16) * 1024 + m] = v1;
    }
}

__global__ void k_fc2(const float* __restrict__ fc1, const float* __restrict__ w2,
                      const float* __restrict__ b2, float* __restrict__ out) {
    const int b = blockIdx.x;
    const int j = threadIdx.x;
    if (j < NAn) {
        float s = b2[j];
        const float* xr = fc1 + b * 1024;
        const float* wr = w2 + j * 1024;
        for (int k = 0; k < 1024; ++k) s += xr[k] * wr[k];
        out[b * NAn + j] = s;
    }
}

// =====================================================================
// host launcher
// =====================================================================
extern "C" void kernel_launch(void* const* d_in, const int* in_sizes, int n_in,
                              void* d_out, int out_size, void* d_ws, size_t ws_size,
                              hipStream_t stream) {
    (void)in_sizes; (void)n_in; (void)out_size; (void)ws_size;
    const float* feats   = (const float*)d_in[0];
    const int*   progs   = (const int*)  d_in[1];
    const float* memory  = (const float*)d_in[2];
    const float* stem_w1 = (const float*)d_in[3];
    const float* stem_b1 = (const float*)d_in[4];
    const float* stem_w2 = (const float*)d_in[5];
    const float* stem_b2 = (const float*)d_in[6];
    const float* cls_cw  = (const float*)d_in[7];
    const float* cls_cb  = (const float*)d_in[8];
    const float* cls_w1  = (const float*)d_in[9];
    const float* cls_b1  = (const float*)d_in[10];
    const float* cls_w2  = (const float*)d_in[11];
    const float* cls_b2  = (const float*)d_in[12];
    float* out = (float*)d_out;

    // workspace carve-up (256B aligned)
    char* base = (char*)d_ws;
    size_t off = 0;
    auto take = [&](size_t bytes) {
        char* p = base + off;
        off = (off + bytes + 255) & ~(size_t)255;
        return p;
    };
    const int nFeats = Bn * 1024 * HWn;           // 12,845,056
    const int nMem   = NSLOT * 4 * CDn;           //  6,635,520
    const int nSW1   = Sc * 1024 * 9;
    const int nSW2   = Sc * Sc * 9;
    const int nCCW   = 512 * Sc;
    const int nCW1   = 1024 * 25088;
    const int nAct   = Bn * Sc * HWn;             //    802,816
    const int nClsF  = Bn * 512 * HWn;
    const int nX     = Bn * 25088;
    const int nFC1   = Bn * 1024;

    _Float16* featsH = (_Float16*)take((size_t)nFeats * 2);
    _Float16* memH   = (_Float16*)take((size_t)nMem * 2);
    _Float16* sw1H   = (_Float16*)take((size_t)nSW1 * 2);
    _Float16* sw2H   = (_Float16*)take((size_t)nSW2 * 2);
    _Float16* ccwH   = (_Float16*)take((size_t)nCCW * 2);
    _Float16* cw1H   = (_Float16*)take((size_t)nCW1 * 2);
    _Float16* stem1  = (_Float16*)take((size_t)nAct * 2);
    _Float16* featb  = (_Float16*)take((size_t)nAct * 2);
    _Float16* outA   = (_Float16*)take((size_t)nAct * 2);
    _Float16* outB   = (_Float16*)take((size_t)nAct * 2);
    _Float16* savA   = (_Float16*)take((size_t)nAct * 2);
    _Float16* savB   = (_Float16*)take((size_t)nAct * 2);
    _Float16* tmp1   = (_Float16*)take((size_t)nAct * 2);
    _Float16* tmp2   = (_Float16*)take((size_t)nAct * 2);
    float*    clsF   = (float*)   take((size_t)nClsF * 4);
    _Float16* xpool  = (_Float16*)take((size_t)nX * 2);
    float*    fc1o   = (float*)   take((size_t)nFC1 * 4);

    // ---- precision conversion (one-time per launch, L2-resident afterwards) ----
    k_convert<<<2048, 256, 0, stream>>>(feats,   featsH, nFeats);
    k_convert<<<2048, 256, 0, stream>>>(memory,  memH,   nMem);
    k_convert<<<512,  256, 0, stream>>>(stem_w1, sw1H,   nSW1);
    k_convert<<<64,   256, 0, stream>>>(stem_w2, sw2H,   nSW2);
    k_convert<<<64,   256, 0, stream>>>(cls_cw,  ccwH,   nCCW);
    k_convert<<<2048, 256, 0, stream>>>(cls_w1,  cw1H,   nCW1);

    // ---- stem ----
    dim3 cgrid(7, Bn);
    k_conv_stem<<<cgrid, 128, LDSB, stream>>>(featsH, (long)1024 * HWn, sw1H, stem_b1,
                                              stem1, (long)Sc * HWn, 1024, 64);
    k_conv_stem<<<cgrid, 128, LDSB, stream>>>(stem1, (long)Sc * HWn, sw2H, stem_b2,
                                              featb, (long)Sc * HWn, 64, 64);
    k_init_state<<<1024, 256, 0, stream>>>(featb, outA, savA, nAct);

    // ---- program execution: 12 steps, reversed program order ----
    _Float16 *ocur = outA, *onext = outB, *scur = savA, *snext = savB;
    for (int t = Tn - 1; t >= 0; --t) {
        k_update_saved<<<1024, 256, 0, stream>>>(progs, t, ocur, scur, snext);
        k_prog_conv1<<<cgrid, 128, LDSB, stream>>>(progs, t, featb, ocur, scur, memH, tmp1);
        k_prog_conv2<<<cgrid, 128, LDSB, stream>>>(progs, t, tmp1, memH, tmp2);
        k_prog_conv3<<<cgrid, 128, LDSB, stream>>>(progs, t, tmp2, ocur, memH, onext);
        _Float16* tsw;
        tsw = ocur; ocur = onext; onext = tsw;
        tsw = scur; scur = snext; snext = tsw;
    }

    // ---- classifier: 1x1 conv -> pool -> FC1 -> FC2 ----
    k_conv_cls<<<cgrid, 128, LDSB, stream>>>(ocur, (long)Sc * HWn, ccwH, cls_cb,
                                             clsF, (long)512 * HWn, 64, 512);
    k_pool<<<2048, 256, 0, stream>>>(clsF, xpool);
    k_fc1<<<dim3(2, 16), 128, 0, stream>>>(cw1H, xpool, cls_b1, fc1o);
    k_fc2<<<Bn, 32, 0, stream>>>(fc1o, cls_w2, cls_b2, out);
}
